// Qwen3SelectiveModel_81784767250712
// MI455X (gfx1250) — matrix operand order repaired and checked by reading, not compiled, not run on hardware
//
#include <hip/hip_runtime.h>

// ---------------- constants (match reference) ----------------
#define T_   8192
#define Hd   2048
#define Lc   2
#define NHh  16
#define NKVh 8
#define DHd  128
#define Fd   8192
#define Bc   8
#define Sc   1024   // T_/Bc

typedef __attribute__((ext_vector_type(16))) __bf16 bf16x16;
typedef __attribute__((ext_vector_type(8)))  __bf16 bf16x8;
typedef __attribute__((ext_vector_type(8)))  float  f32x8;

union F8   { f32x8 v; float f[8]; };
union BF16 { bf16x16 v; bf16x8 h[2]; };

static __device__ inline f32x8 wmma_bf16(bf16x16 a, bf16x16 b, f32x8 c) {
  return __builtin_amdgcn_wmma_f32_16x16x32_bf16(false, a, false, b, (short)0, c, false, false);
}

// --------- CDNA5 async global->LDS copies (guarded; sync fallback) ----------
#if defined(__AMDGCN__) && defined(__has_builtin)
#  if __has_builtin(__builtin_amdgcn_global_load_async_to_lds_b128) && \
      __has_builtin(__builtin_amdgcn_s_wait_asynccnt)
#    define HAS_ASYNC 1
#  endif
#endif
#ifndef HAS_ASYNC
#  define HAS_ASYNC 0
#endif

#if HAS_ASYNC
#define GLOBAL_AS __attribute__((address_space(1)))
#define LDS_AS    __attribute__((address_space(3)))
typedef __attribute__((vector_size(4 * sizeof(int)))) int i32x4_t;
static __device__ inline void async_cp16(const __bf16* g, __bf16* l) {
  __builtin_amdgcn_global_load_async_to_lds_b128(
      (GLOBAL_AS i32x4_t*)(unsigned long long)(size_t)g,
      (LDS_AS i32x4_t*)(unsigned)(size_t)l, 0, 0);
}
#define WAIT_ASYNC(n) __builtin_amdgcn_s_wait_asynccnt(n)
#else
static __device__ inline void async_cp16(const __bf16* g, __bf16* l) {
  *(bf16x8*)l = *(const bf16x8*)g;  // sync fallback (compiler handles waits)
}
#define WAIT_ASYNC(n) ((void)0)
#endif

// ---------------- RMSNorm over H columns ----------------
__global__ __launch_bounds__(256)
void rmsnorm_k(const float* __restrict__ x, const float* __restrict__ w,
               __bf16* outB, float* outF, int ncols) {
  int t = blockIdx.x;
  const float* row = x + (size_t)t * ncols;
  float ss = 0.f;
  for (int i = threadIdx.x; i < ncols; i += 256) { float v = row[i]; ss = fmaf(v, v, ss); }
#pragma unroll
  for (int m = 16; m; m >>= 1) ss += __shfl_xor(ss, m);
  __shared__ float red[8];
  __shared__ float sInv;
  if ((threadIdx.x & 31) == 0) red[threadIdx.x >> 5] = ss;
  __syncthreads();
  if (threadIdx.x == 0) {
    float s = 0.f;
    for (int i = 0; i < 8; i++) s += red[i];
    sInv = rsqrtf(s / (float)ncols + 1e-6f);
  }
  __syncthreads();
  float inv = sInv;
  for (int i = threadIdx.x; i < ncols; i += 256) {
    float v = row[i] * inv * w[i];
    if (outB) outB[(size_t)t * ncols + i] = (__bf16)v;
    else      outF[(size_t)t * ncols + i] = v;
  }
}

// ------- per-head RMSNorm + RoPE, relayout to [B][nh][S][DH] bf16 -----------
__global__ __launch_bounds__(128)
void qknorm_rope(const float* __restrict__ in, const float* __restrict__ nw,
                 const int* __restrict__ pids, __bf16* __restrict__ out, int nh) {
  int h = blockIdx.x, t = blockIdx.y, i = threadIdx.x;
  float v = in[((size_t)t * nh + h) * DHd + i];
  float ss = v * v;
#pragma unroll
  for (int m = 16; m; m >>= 1) ss += __shfl_xor(ss, m);
  __shared__ float red[4];
  __shared__ float sInv;
  __shared__ float sh[DHd];
  if ((threadIdx.x & 31) == 0) red[threadIdx.x >> 5] = ss;
  __syncthreads();
  if (threadIdx.x == 0)
    sInv = rsqrtf((red[0] + red[1] + red[2] + red[3]) / (float)DHd + 1e-6f);
  __syncthreads();
  float xn = v * sInv * nw[i];
  sh[i] = xn;
  __syncthreads();
  float partner = (i < 64) ? sh[i + 64] : sh[i - 64];
  int j = i & 63;
  float invf = __expf(-((float)(2 * j) / 128.0f) * logf(1.0e6f));
  float pos  = (float)(pids[t] % Sc);
  float ang  = pos * invf;
  float c = __cosf(ang), s = __sinf(ang);
  float rot = (i < 64) ? -partner : partner;
  float res = xn * c + rot * s;
  int b = t / Sc, sI = t % Sc;
  out[(((size_t)b * nh + h) * Sc + sI) * DHd + i] = (__bf16)res;
}

// ------ V relayout [T][NKV][DH] f32 -> [B][NKV][DH][S] bf16 (dh-major) ------
__global__ __launch_bounds__(128)
void v_relayout(const float* __restrict__ vin, __bf16* __restrict__ vout) {
  int h = blockIdx.x, t = blockIdx.y, i = threadIdx.x;
  int b = t / Sc, sI = t % Sc;
  vout[(((size_t)b * NKVh + h) * DHd + i) * Sc + sI] =
      (__bf16)vin[((size_t)t * NKVh + h) * DHd + i];
}

// ------ fp32 [K][N] -> bf16 transposed [N][K] (LDS-tiled, coalesced) --------
__global__ __launch_bounds__(256)
void cvt_transpose_bf16(const float* __restrict__ src, __bf16* __restrict__ dst,
                        int K, int N) {
  __shared__ float tile[32][33];
  int n0 = blockIdx.x * 32, k0 = blockIdx.y * 32;
  int tx = threadIdx.x & 31, ty = threadIdx.x >> 5;
#pragma unroll
  for (int i = ty; i < 32; i += 8)
    tile[i][tx] = src[(size_t)(k0 + i) * N + n0 + tx];
  __syncthreads();
#pragma unroll
  for (int i = ty; i < 32; i += 8)
    dst[(size_t)(n0 + i) * K + k0 + tx] = (__bf16)tile[tx][i];
}

// ------- tiled bf16 WMMA GEMM: C[MxN] = A[MxK] * BT[NxK]^T ------------------
// Double-buffered LDS, async global->LDS staging.
// mode 0: f32 out (+ optional addend); 1: bf16 silu(acc); 2: bf16 acc*partner
__global__ __launch_bounds__(256)
void gemm_bf16_wmma(const __bf16* __restrict__ A, const __bf16* __restrict__ BT,
                    int M, int N, int K,
                    float* outF, __bf16* outB,
                    const float* addend, const __bf16* partner, int mode) {
  __shared__ __align__(16) __bf16 sA[2][128][40];  // [buf][row][k]
  __shared__ __align__(16) __bf16 sB[2][128][40];  // [buf][col][k]

  int m0 = blockIdx.y * 128;
  int n0 = blockIdx.x * 128;
  int tid = threadIdx.x;
  int lane = tid & 31;
  int wave = tid >> 5;
  int wm = wave >> 1;
  int wn = wave & 1;

  F8 acc[2][4];
#pragma unroll
  for (int a = 0; a < 2; a++)
#pragma unroll
    for (int b = 0; b < 4; b++)
#pragma unroll
      for (int e = 0; e < 8; e++) acc[a][b].f[e] = 0.f;

  int frow = lane & 15;
  int aoff = (lane < 16) ? 0 : 8;
  int kst  = (lane < 16) ? 0 : 8;

  // staging map: 256 threads, each copies 2x16B for A and 2x16B for B per tile
  int srow = tid >> 1;
  int sseg = (tid & 1) * 16;
  const __bf16* gA = A  + (size_t)(m0 + srow) * K + sseg;
  const __bf16* gB = BT + (size_t)(n0 + srow) * K + sseg;

  auto stage = [&](int k0, int buf) {
    async_cp16(gA + k0,     &sA[buf][srow][sseg]);
    async_cp16(gA + k0 + 8, &sA[buf][srow][sseg + 8]);
    async_cp16(gB + k0,     &sB[buf][srow][sseg]);
    async_cp16(gB + k0 + 8, &sB[buf][srow][sseg + 8]);
  };

  int nk = K >> 5;
  stage(0, 0);
  for (int ki = 0; ki < nk; ki++) {
    int buf = ki & 1;
    if (ki + 1 < nk) {
      stage((ki + 1) << 5, buf ^ 1);
      WAIT_ASYNC(4);          // previous tile complete; 4 new in flight
    } else {
      WAIT_ASYNC(0);
    }
    __syncthreads();

    BF16 af[2], bfr[4];
#pragma unroll
    for (int a = 0; a < 2; a++) {
      const __bf16* pA = &sA[buf][wm * 32 + a * 16 + frow][0];
      af[a].h[0] = *(const bf16x8*)(pA + aoff);
      af[a].h[1] = *(const bf16x8*)(pA + aoff + 16);
    }
#pragma unroll
    for (int b = 0; b < 4; b++) {
      const __bf16* pB = &sB[buf][wn * 64 + b * 16 + frow][0];
      bfr[b].h[0] = *(const bf16x8*)(pB + kst);
      bfr[b].h[1] = *(const bf16x8*)(pB + kst + 16);
    }
#pragma unroll
    for (int a = 0; a < 2; a++)
#pragma unroll
      for (int b = 0; b < 4; b++)
        acc[a][b].v = wmma_bf16(af[a].v, bfr[b].v, acc[a][b].v);
    __syncthreads();
  }

  int rowsel = (lane < 16) ? 0 : 8;
#pragma unroll
  for (int a = 0; a < 2; a++)
#pragma unroll
    for (int b = 0; b < 4; b++)
#pragma unroll
      for (int e = 0; e < 8; e++) {
        int row = m0 + wm * 32 + a * 16 + rowsel + e;
        int col = n0 + wn * 64 + b * 16 + (lane & 15);
        size_t idx = (size_t)row * N + col;
        float v = acc[a][b].f[e];
        if (mode == 0) {
          if (addend) v += addend[idx];
          outF[idx] = v;
        } else if (mode == 1) {
          outB[idx] = (__bf16)(v / (1.f + __expf(-v)));
        } else {
          outB[idx] = (__bf16)(v * (float)partner[idx]);
        }
      }
}

// ---------------- fused causal GQA flash attention (WMMA) --------------------
// Q: [B][NH][S][DH], K: [B][NKV][S][DH], V: [B][NKV][DH][S], out: [T][NH][DH]
__global__ __launch_bounds__(128)
void attn_fused(const __bf16* __restrict__ Q, const __bf16* __restrict__ Kt,
                const __bf16* __restrict__ Vt, __bf16* __restrict__ O) {
  __shared__ __align__(16) __bf16 sK[2][32][136];   // [buf][key][dh]
  __shared__ __align__(16) __bf16 sV[2][128][40];   // [buf][dh][key]
  __shared__ __align__(16) __bf16 sP[4][16][40];    // per-wave P tiles

  int qb = blockIdx.x, h = blockIdx.y, b = blockIdx.z;
  int tid = threadIdx.x, lane = tid & 31, w = tid >> 5;
  int kvh = h >> 1;  // NH/NKV = 2

  const __bf16* qbase = Q + (((size_t)b * NHh + h) * Sc + qb * 64 + w * 16) * DHd;
  const __bf16* kbase = Kt + ((size_t)b * NKVh + kvh) * Sc * DHd;
  const __bf16* vbase = Vt + ((size_t)b * NKVh + kvh) * (size_t)DHd * Sc;

  int frow = lane & 15;
  int aoff = (lane < 16) ? 0 : 8;
  int kst  = (lane < 16) ? 0 : 8;

  // Q fragments (16 rows x 128 dh) stay in registers for the whole kernel
  BF16 qf[4];
#pragma unroll
  for (int c = 0; c < 4; c++) {
    const __bf16* p = qbase + (size_t)frow * DHd + c * 32;
    qf[c].h[0] = *(const bf16x8*)(p + aoff);
    qf[c].h[1] = *(const bf16x8*)(p + aoff + 16);
  }

  // staging map: 128 threads; K: 4x16B per thread, V: 4x16B per thread
  int skey = tid >> 2;
  int sdseg = (tid & 3) * 32;
  const __bf16* kg = kbase + (size_t)skey * DHd + sdseg;
  const __bf16* vg = vbase + (size_t)tid * Sc;  // dh = tid

  auto stageKV = [&](int kt2, int buf) {
    int k0 = kt2 * 32;
#pragma unroll
    for (int j = 0; j < 4; j++)
      async_cp16(kg + (size_t)k0 * DHd + j * 8, &sK[buf][skey][sdseg + j * 8]);
#pragma unroll
    for (int j = 0; j < 4; j++)
      async_cp16(vg + k0 + j * 8, &sV[buf][tid][j * 8]);
  };

  F8 o[8];
  float mrow[8], lrow[8];
#pragma unroll
  for (int ch = 0; ch < 8; ch++)
#pragma unroll
    for (int e = 0; e < 8; e++) o[ch].f[e] = 0.f;
#pragma unroll
  for (int e = 0; e < 8; e++) { mrow[e] = -1e30f; lrow[e] = 0.f; }

  const float scale = 0.08838834764831845f;  // 1/sqrt(128)
  int qrow_base = qb * 64 + w * 16;
  int numKT = 2 * (qb + 1);

  stageKV(0, 0);
  for (int kt = 0; kt < numKT; kt++) {
    int buf = kt & 1;
    int k0 = kt * 32;
    if (kt + 1 < numKT) {
      stageKV(kt + 1, buf ^ 1);
      WAIT_ASYNC(8);
    } else {
      WAIT_ASYNC(0);
    }
    __syncthreads();

    // ---- scores: two 16x16 tiles (32 keys), K-dim = DH via 4 chunks ----
    F8 sc[2];
#pragma unroll
    for (int n = 0; n < 2; n++) {
#pragma unroll
      for (int e = 0; e < 8; e++) sc[n].f[e] = 0.f;
#pragma unroll
      for (int c = 0; c < 4; c++) {
        BF16 bk;
        const __bf16* pB = &sK[buf][n * 16 + frow][c * 32 + kst];
        bk.h[0] = *(const bf16x8*)pB;
        bk.h[1] = *(const bf16x8*)(pB + 16);
        sc[n].v = wmma_bf16(qf[c].v, bk.v, sc[n].v);
      }
    }

    // ---- online softmax update ----
#pragma unroll
    for (int e = 0; e < 8; e++) {
      int row  = qrow_base + ((lane < 16) ? e : 8 + e);
      int col0 = k0 + (lane & 15);
      float s0 = sc[0].f[e] * scale;
      float s1 = sc[1].f[e] * scale;
      if (col0 > row)      s0 = -1e30f;
      if (col0 + 16 > row) s1 = -1e30f;
      float mt = fmaxf(s0, s1);
#pragma unroll
      for (int msk = 8; msk >= 1; msk >>= 1) mt = fmaxf(mt, __shfl_xor(mt, msk));
      float mnew = fmaxf(mrow[e], mt);
      float corr = __expf(mrow[e] - mnew);
      float p0 = __expf(s0 - mnew);
      float p1 = __expf(s1 - mnew);
      float rs = p0 + p1;
#pragma unroll
      for (int msk = 8; msk >= 1; msk >>= 1) rs += __shfl_xor(rs, msk);
      lrow[e] = lrow[e] * corr + rs;
      mrow[e] = mnew;
#pragma unroll
      for (int ch = 0; ch < 8; ch++) o[ch].f[e] *= corr;
      int prow = (lane < 16) ? e : 8 + e;
      sP[w][prow][lane & 15]        = (__bf16)p0;
      sP[w][prow][16 + (lane & 15)] = (__bf16)p1;
    }
    __syncthreads();

    // ---- PV: A = P (16x32), B = V^T tile, 8 dh chunks ----
    BF16 pf;
    {
      const __bf16* pp = &sP[w][frow][0];
      pf.h[0] = *(const bf16x8*)(pp + aoff);
      pf.h[1] = *(const bf16x8*)(pp + aoff + 16);
    }
#pragma unroll
    for (int ch = 0; ch < 8; ch++) {
      BF16 bv;
      const __bf16* pv = &sV[buf][ch * 16 + frow][kst];
      bv.h[0] = *(const bf16x8*)pv;
      bv.h[1] = *(const bf16x8*)(pv + 16);
      o[ch].v = wmma_bf16(pf.v, bv.v, o[ch].v);
    }
    __syncthreads();
  }

  // ---- finalize ----
#pragma unroll
  for (int e = 0; e < 8; e++) {
    float inv = 1.0f / lrow[e];
    int t = b * Sc + qrow_base + ((lane < 16) ? e : 8 + e);
#pragma unroll
    for (int ch = 0; ch < 8; ch++) {
      int dh = ch * 16 + (lane & 15);
      O[((size_t)t * NHh + h) * DHd + dh] = (__bf16)(o[ch].f[e] * inv);
    }
  }
}

// ---------------- host orchestration ----------------
static constexpr size_t NQW = (size_t)Lc * Hd * NHh * DHd;
static constexpr size_t NKW = (size_t)Lc * Hd * NKVh * DHd;
static constexpr size_t NVW = NKW;
static constexpr size_t NOW = NQW;
static constexpr size_t NGW = (size_t)Lc * Hd * Fd;
static constexpr size_t NUW = NGW;
static constexpr size_t NDW = NGW;

extern "C" void kernel_launch(void* const* d_in, const int* in_sizes, int n_in,
                              void* d_out, int out_size, void* d_ws, size_t ws_size,
                              hipStream_t stream) {
  (void)in_sizes; (void)n_in; (void)out_size; (void)ws_size;

  const float* hidden = (const float*)d_in[0];
  const int*   pids   = (const int*)d_in[1];
  const float* ln1    = (const float*)d_in[3];
  const float* qw     = (const float*)d_in[4];
  const float* kw     = (const float*)d_in[5];
  const float* vw     = (const float*)d_in[6];
  const float* ow     = (const float*)d_in[7];
  const float* qnw    = (const float*)d_in[8];
  const float* knw    = (const float*)d_in[9];
  const float* ln2    = (const float*)d_in[10];
  const float* gw     = (const float*)d_in[11];
  const float* uw     = (const float*)d_in[12];
  const float* dw     = (const float*)d_in[13];
  const float* finw   = (const float*)d_in[14];

  char* p = (char*)d_ws;
  auto alloc = [&](size_t bytes) -> char* {
    char* r = p;
    p += (bytes + 255) & ~(size_t)255;
    return r;
  };

  __bf16* wqb = (__bf16*)alloc(NQW * 2);   // all weights stored TRANSPOSED [N][K]
  __bf16* wkb = (__bf16*)alloc(NKW * 2);
  __bf16* wvb = (__bf16*)alloc(NVW * 2);
  __bf16* wob = (__bf16*)alloc(NOW * 2);
  __bf16* wgb = (__bf16*)alloc(NGW * 2);
  __bf16* wub = (__bf16*)alloc(NUW * 2);
  __bf16* wdb = (__bf16*)alloc(NDW * 2);

  float*  xbuf = (float*)alloc((size_t)T_ * Hd * 4);
  __bf16* hbf  = (__bf16*)alloc((size_t)T_ * Hd * 2);
  float*  qf32 = (float*)alloc((size_t)T_ * NHh * DHd * 4);
  float*  kf32 = (float*)alloc((size_t)T_ * NKVh * DHd * 4);
  float*  vf32 = (float*)alloc((size_t)T_ * NKVh * DHd * 4);
  __bf16* qh   = (__bf16*)alloc((size_t)T_ * NHh * DHd * 2);
  __bf16* kh   = (__bf16*)alloc((size_t)T_ * NKVh * DHd * 2);
  __bf16* vh   = (__bf16*)alloc((size_t)T_ * NKVh * DHd * 2);
  __bf16* aout = (__bf16*)alloc((size_t)T_ * NHh * DHd * 2);
  __bf16* gact = (__bf16*)alloc((size_t)T_ * Fd * 2);
  __bf16* mact = (__bf16*)alloc((size_t)T_ * Fd * 2);

  // weight conversion + transpose (once per call; streaming prologue)
  auto cvtT = [&](const float* src, __bf16* dst, int K, int N) {
    for (int l = 0; l < Lc; l++)
      cvt_transpose_bf16<<<dim3(N / 32, K / 32), 256, 0, stream>>>(
          src + (size_t)l * K * N, dst + (size_t)l * N * K, K, N);
  };
  cvtT(qw, wqb, Hd, NHh * DHd);
  cvtT(kw, wkb, Hd, NKVh * DHd);
  cvtT(vw, wvb, Hd, NKVh * DHd);
  cvtT(ow, wob, NHh * DHd, Hd);
  cvtT(gw, wgb, Hd, Fd);
  cvtT(uw, wub, Hd, Fd);
  cvtT(dw, wdb, Fd, Hd);

  (void)hipMemcpyAsync(xbuf, hidden, (size_t)T_ * Hd * 4, hipMemcpyDeviceToDevice, stream);

  auto gemm = [&](const __bf16* A, const __bf16* BT, int M, int N, int K,
                  float* oF, __bf16* oB, const float* add, const __bf16* part, int mode) {
    gemm_bf16_wmma<<<dim3(N / 128, M / 128), 256, 0, stream>>>(A, BT, M, N, K, oF, oB, add, part, mode);
  };

  for (int l = 0; l < Lc; l++) {
    const __bf16* wq_l = wqb + (size_t)l * Hd * NHh * DHd;
    const __bf16* wk_l = wkb + (size_t)l * Hd * NKVh * DHd;
    const __bf16* wv_l = wvb + (size_t)l * Hd * NKVh * DHd;
    const __bf16* wo_l = wob + (size_t)l * NHh * DHd * Hd;
    const __bf16* wg_l = wgb + (size_t)l * Hd * Fd;
    const __bf16* wu_l = wub + (size_t)l * Hd * Fd;
    const __bf16* wd_l = wdb + (size_t)l * Fd * Hd;

    // attention half
    rmsnorm_k<<<T_, 256, 0, stream>>>(xbuf, ln1 + (size_t)l * Hd, hbf, nullptr, Hd);
    gemm(hbf, wq_l, T_, NHh * DHd, Hd, qf32, nullptr, nullptr, nullptr, 0);
    gemm(hbf, wk_l, T_, NKVh * DHd, Hd, kf32, nullptr, nullptr, nullptr, 0);
    gemm(hbf, wv_l, T_, NKVh * DHd, Hd, vf32, nullptr, nullptr, nullptr, 0);
    qknorm_rope<<<dim3(NHh, T_), 128, 0, stream>>>(qf32, qnw + (size_t)l * DHd, pids, qh, NHh);
    qknorm_rope<<<dim3(NKVh, T_), 128, 0, stream>>>(kf32, knw + (size_t)l * DHd, pids, kh, NKVh);
    v_relayout<<<dim3(NKVh, T_), 128, 0, stream>>>(vf32, vh);
    attn_fused<<<dim3(Sc / 64, NHh, Bc), 128, 0, stream>>>(qh, kh, vh, aout);
    gemm(aout, wo_l, T_, Hd, NHh * DHd, xbuf, nullptr, xbuf, nullptr, 0);  // + residual

    // MLP half
    rmsnorm_k<<<T_, 256, 0, stream>>>(xbuf, ln2 + (size_t)l * Hd, hbf, nullptr, Hd);
    gemm(hbf, wg_l, T_, Fd, Hd, nullptr, gact, nullptr, nullptr, 1);       // silu(gate)
    gemm(hbf, wu_l, T_, Fd, Hd, nullptr, mact, nullptr, gact, 2);          // * up
    gemm(mact, wd_l, T_, Hd, Fd, xbuf, nullptr, xbuf, nullptr, 0);         // + residual
  }

  // final RMSNorm -> fp32 output
  rmsnorm_k<<<T_, 256, 0, stream>>>(xbuf, finw, nullptr, (float*)d_out, Hd);
}
